// LtsmUnit_54546084659583
// MI455X (gfx1250) — matrix operand order repaired and checked by reading, not compiled, not run on hardware
//
#include <hip/hip_runtime.h>
#include <hip/hip_bf16.h>

typedef __attribute__((ext_vector_type(16))) _Float16 v16h;
typedef __attribute__((ext_vector_type(8)))  _Float16 v8h;
typedef __attribute__((ext_vector_type(8)))  float    v8f;
typedef __attribute__((ext_vector_type(4)))  float    v4f;

__device__ __forceinline__ float fast_sigmoid(float x) {
    // 1 / (1 + 2^(-x*log2 e))  -- v_exp_f32 + v_rcp_f32, co-executes with WMMA
    return __builtin_amdgcn_rcpf(1.0f + __builtin_amdgcn_exp2f(-1.4426950408889634f * x));
}
__device__ __forceinline__ float fast_tanh(float x) {
    return 2.0f * fast_sigmoid(2.0f * x) - 1.0f;
}

// One-time weight conversion: stacked gate weights [4*128, 256] and w_out
// [128,128] fp32 -> f16 row-major in workspace (L2/WGP$-resident thereafter).
__global__ __launch_bounds__(256) void prep_weights(
    const float* __restrict__ w,  const float* __restrict__ wi,
    const float* __restrict__ wf, const float* __restrict__ wo,
    const float* __restrict__ w_out,
    _Float16* __restrict__ Wh, _Float16* __restrict__ Woh) {
    int i = blockIdx.x * blockDim.x + threadIdx.x;   // 0 .. 131071
    int g = i >> 15;                                 // which gate matrix
    int r = i & 32767;
    const float* src = (g == 0) ? w : (g == 1) ? wi : (g == 2) ? wf : wo;
    Wh[i] = (_Float16)src[r];
    if (i < 128 * 128) Woh[i] = (_Float16)w_out[i];
}

// Fused LSTM cell. Each wave32 owns 16 rows; block of 8 waves owns 128 rows.
__global__ __launch_bounds__(256) void lstm_fused(
    const float* __restrict__ c_prev, const float* __restrict__ h_prev,
    const float* __restrict__ x,
    const _Float16* __restrict__ Wh,   // [512][256] f16 (gates stacked z,i,f,o)
    const _Float16* __restrict__ Woh,  // [128][128] f16
    const float* __restrict__ b,  const float* __restrict__ bi,
    const float* __restrict__ bf, const float* __restrict__ bo,
    const float* __restrict__ b_out,
    float* __restrict__ out_c, float* __restrict__ out_h,
    float* __restrict__ out_y) {

    __shared__ _Float16 hbuf[8 * 16 * 128];          // 32 KB: per-wave h tile (f16)

    const int lane    = threadIdx.x & 31;
    const int wave    = threadIdx.x >> 5;
    const int rowbase = blockIdx.x * 128 + wave * 16;
    const int lm      = lane & 15;                   // row (A) / column (B,D) within tile
    const int hgrp    = lane >> 4;                   // 0: lanes 0-15, 1: lanes 16-31
    const int ka      = hgrp * 8;                    // unused; A fragments also load at kb
    const int kb      = hgrp * 16;                   // B-fragment K offset (halves)
    const int rsel    = hgrp * 8;                    // D-fragment row offset

    // ---- A fragments: xh[16 rows][256] fp32 -> f16, contiguous k = 16*hgrp + s (same as B) ----
    const int arow = rowbase + lm;
    v16h afrag[8];
    #pragma unroll
    for (int ks = 0; ks < 8; ++ks) {
        const float* src = (ks < 4) ? (x      + (size_t)arow * 128 + ks * 32)
                                    : (h_prev + (size_t)arow * 128 + (ks - 4) * 32);
        v4f f0 = *(const v4f*)(src + kb);
        v4f f1 = *(const v4f*)(src + kb + 4);
        v4f f2 = *(const v4f*)(src + kb + 8);
        v4f f3 = *(const v4f*)(src + kb + 12);
        v16h a;
        #pragma unroll
        for (int j = 0; j < 4; ++j) {
            a[j]      = (_Float16)f0[j];
            a[4 + j]  = (_Float16)f1[j];
            a[8 + j]  = (_Float16)f2[j];
            a[12 + j] = (_Float16)f3[j];
        }
        afrag[ks] = a;
    }

    _Float16* hrow = hbuf + wave * (16 * 128);

    // ---- gates GEMM + activations, 16-column tiles; z/i/f/o tiles together ----
    #pragma unroll
    for (int t = 0; t < 8; ++t) {
        const int col = t * 16 + lm;                 // D column for this lane
        v8f az, ai, af, ao;
        const float vz = b[col], vi = bi[col], vf = bf[col], vo = bo[col];
        #pragma unroll
        for (int r = 0; r < 8; ++r) { az[r] = vz; ai[r] = vi; af[r] = vf; ao[r] = vo; }

        const _Float16* wb = Wh + (size_t)(t * 16 + lm) * 256 + kb;

        // Software-pipelined B fragments: issue loads for ks+1 before the
        // four back-to-back WMMAs of step ks (pure renaming after unroll).
        v16h nz = *(const v16h*)(wb + 0u * 32768);
        v16h ni = *(const v16h*)(wb + 1u * 32768);
        v16h nf = *(const v16h*)(wb + 2u * 32768);
        v16h no = *(const v16h*)(wb + 3u * 32768);
        #pragma unroll
        for (int ks = 0; ks < 8; ++ks) {
            v16h cz = nz, ci = ni, cf = nf, co = no;
            if (ks < 7) {
                const _Float16* wk = wb + (ks + 1) * 32;
                nz = *(const v16h*)(wk + 0u * 32768);
                ni = *(const v16h*)(wk + 1u * 32768);
                nf = *(const v16h*)(wk + 2u * 32768);
                no = *(const v16h*)(wk + 3u * 32768);
            }
            az = __builtin_amdgcn_wmma_f32_16x16x32_f16(false, afrag[ks], false, cz, (short)0, az, false, false);
            ai = __builtin_amdgcn_wmma_f32_16x16x32_f16(false, afrag[ks], false, ci, (short)0, ai, false, false);
            af = __builtin_amdgcn_wmma_f32_16x16x32_f16(false, afrag[ks], false, cf, (short)0, af, false, false);
            ao = __builtin_amdgcn_wmma_f32_16x16x32_f16(false, afrag[ks], false, co, (short)0, ao, false, false);
        }

        #pragma unroll
        for (int r = 0; r < 8; ++r) {
            const int    row = rowbase + r + rsel;   // D layout: VGPR r -> rows r / r+8
            const size_t idx = (size_t)row * 128 + col;
            const float  cp  = c_prev[idx];
            const float  z   = fast_tanh(az[r]);
            const float  si  = fast_sigmoid(ai[r]);
            const float  sf  = fast_sigmoid(af[r]);
            const float  so  = fast_sigmoid(ao[r]);
            const float  cn  = sf * cp + si * z;
            const float  hn  = so * fast_tanh(cn);
            out_c[idx] = cn;
            out_h[idx] = hn;
            hrow[(r + rsel) * 128 + col] = (_Float16)hn;  // stage for output GEMM
        }
    }

    // ---- rebuild h as A fragments from LDS (same-wave DS ops are in order) ----
    v16h hfrag[4];
    #pragma unroll
    for (int ks = 0; ks < 4; ++ks) {
        const _Float16* hp = hrow + lm * 128 + ks * 32 + kb;
        v8h lo = *(const v8h*)(hp);
        v8h hi = *(const v8h*)(hp + 8);
        v16h a;
        #pragma unroll
        for (int j = 0; j < 8; ++j) { a[j] = lo[j]; a[8 + j] = hi[j]; }
        hfrag[ks] = a;
    }

    // ---- y = sigmoid(h @ w_out^T + b_out) ----
    #pragma unroll
    for (int t = 0; t < 8; ++t) {
        const int col = t * 16 + lm;
        v8f acc;
        const float bb = b_out[col];
        #pragma unroll
        for (int r = 0; r < 8; ++r) acc[r] = bb;

        const _Float16* ob = Woh + (size_t)col * 128 + kb;
        v16h bn = *(const v16h*)(ob);
        #pragma unroll
        for (int ks = 0; ks < 4; ++ks) {
            v16h bc = bn;
            if (ks < 3) bn = *(const v16h*)(ob + (ks + 1) * 32);
            acc = __builtin_amdgcn_wmma_f32_16x16x32_f16(false, hfrag[ks], false, bc, (short)0, acc, false, false);
        }
        #pragma unroll
        for (int r = 0; r < 8; ++r) {
            const int row = rowbase + r + rsel;
            out_y[(size_t)row * 128 + col] = fast_sigmoid(acc[r]);
        }
    }
}

extern "C" void kernel_launch(void* const* d_in, const int* in_sizes, int n_in,
                              void* d_out, int out_size, void* d_ws, size_t ws_size,
                              hipStream_t stream) {
    const float* c_prev = (const float*)d_in[0];
    const float* h_prev = (const float*)d_in[1];
    const float* x      = (const float*)d_in[2];
    const float* w      = (const float*)d_in[3];
    const float* wi     = (const float*)d_in[4];
    const float* wf     = (const float*)d_in[5];
    const float* wo     = (const float*)d_in[6];
    const float* w_out  = (const float*)d_in[7];
    const float* b      = (const float*)d_in[8];
    const float* bi     = (const float*)d_in[9];
    const float* bf     = (const float*)d_in[10];
    const float* bo     = (const float*)d_in[11];
    const float* b_out  = (const float*)d_in[12];

    const int B = in_sizes[0] / 128;

    _Float16* Wh  = (_Float16*)d_ws;            // 512*256 halves = 256 KB
    _Float16* Woh = Wh + 512 * 256;             // 128*128 halves =  32 KB

    prep_weights<<<512, 256, 0, stream>>>(w, wi, wf, wo, w_out, Wh, Woh);

    float* out_c = (float*)d_out;
    float* out_h = out_c + (size_t)B * 128;
    float* out_y = out_h + (size_t)B * 128;

    lstm_fused<<<B / 128, 256, 0, stream>>>(c_prev, h_prev, x, Wh, Woh,
                                            b, bi, bf, bo, b_out,
                                            out_c, out_h, out_y);
}
